// Graphormer_72541997630016
// MI455X (gfx1250) — matrix operand order repaired and checked by reading, not compiled, not run on hardware
//
#include <hip/hip_runtime.h>
#include <math.h>

typedef __bf16 bf16;
typedef __attribute__((ext_vector_type(8)))  bf16  bf16x8;
typedef __attribute__((ext_vector_type(16))) bf16  bf16x16;
typedef __attribute__((ext_vector_type(8)))  float f32x8;
typedef unsigned int u32;
typedef unsigned long long u64;
typedef u32 u32x4 __attribute__((ext_vector_type(4)));
typedef int i32x4 __attribute__((ext_vector_type(4)));
typedef int i32x8 __attribute__((ext_vector_type(8)));

// ---------- WMMA helpers (CDNA5 gfx1250, wave32) ----------
// A fragment (16x32 bf16): lane r = lane%16 is row M, g = lane/16 selects K halves.
// element e -> k = 16g+e (callers pass lo = base+16g, hi = lo+8)  => two contiguous 16B loads.
__device__ inline bf16x16 frag_from_pair(const bf16* lo, const bf16* hi) {
  bf16x8 a = *(const bf16x8*)lo;
  bf16x8 b = *(const bf16x8*)hi;
  bf16x16 r;
#pragma unroll
  for (int i = 0; i < 8; ++i) { r[i] = a[i]; r[i + 8] = b[i]; }
  return r;
}
// B fragment (32x16 bf16): lane n = lane%16 is column, k = e + 16*(lane/16).
__device__ inline bf16x16 frag_contig(const bf16* p) {
  return *(const bf16x16*)p;
}
__device__ inline f32x8 wmma_bf16(bf16x16 a, bf16x16 b, f32x8 c) {
  return __builtin_amdgcn_wmma_f32_16x16x32_bf16(false, a, false, b, (short)0, c,
                                                 false, false);
}

// ---------- Tensor Data Mover: 2D tile (tile_d0 x tile_d1 elements, 2B each)
// from row-major global (row stride = stride0 elements) into LDS at lds_off.
// D# built per CDNA5 ISA; groups 2/3 zero (2D).
__device__ inline void tdm_load_2d(u32 lds_off, const void* gptr,
                                   u32 tile_d0, u32 tile_d1,
                                   u32 tens_d0, u32 tens_d1, u32 stride0) {
  u64 ga = (u64)(size_t)gptr;
  u32x4 g0;
  g0[0] = 1u;                                           // count=1, user D#
  g0[1] = lds_off;                                      // lds_addr (bytes)
  g0[2] = (u32)ga;                                      // global_addr[31:0]
  g0[3] = (u32)((ga >> 32) & 0x1ffffffu) | (2u << 30);  // addr[56:32] | type=2
  i32x8 g1;
  g1[0] = (int)(1u << 16);                              // data_size=1 -> 2 bytes
  g1[1] = (int)((tens_d0 & 0xffffu) << 16);             // tensor_dim0[15:0]
  g1[2] = (int)(((tens_d0 >> 16) & 0xffffu) |           // tensor_dim0[31:16]
                ((tens_d1 & 0xffffu) << 16));           // tensor_dim1[15:0]
  g1[3] = (int)(((tens_d1 >> 16) & 0xffffu) |           // tensor_dim1[31:16]
                ((tile_d0 & 0xffffu) << 16));           // tile_dim0
  g1[4] = (int)(tile_d1 & 0xffffu);                     // tile_dim1 (tile_dim2=0)
  g1[5] = (int)stride0;                                 // tensor_dim0_stride[31:0]
  g1[6] = 0;                                            // stride hi / dim1 stride
  g1[7] = 0;
  i32x4 z = {0, 0, 0, 0};
  asm volatile("tensor_load_to_lds %0, %1, %2, %3"
               :: "s"(g0), "s"(g1), "s"(z), "s"(z)
               : "memory");
}

// ---------- fp32 weight -> bf16, swizzled into B-fragment order ----------
__global__ void k_swizzle(const float* __restrict__ W, bf16* __restrict__ dst,
                          int K, int N, int tilesTotal, int tileOffset) {
  long idx = (long)blockIdx.x * blockDim.x + threadIdx.x;
  if (idx >= (long)K * N) return;
  int k = (int)(idx / N), n = (int)(idx % N);
  int ck = k >> 5, kk = k & 31;
  int tn = n >> 4, nn = n & 15;
  int g = kk >> 4, e = kk & 15;
  int lane = nn + 16 * g;
  size_t o = (((size_t)ck * tilesTotal + tileOffset + tn) * 32 + lane) * 16 + e;
  dst[o] = (bf16)W[idx];
}

__global__ void k_pack_qkv_bias(const float* __restrict__ bq,
                                const float* __restrict__ bk,
                                const float* __restrict__ bv,
                                float* __restrict__ dst) {
  int idx = blockIdx.x * blockDim.x + threadIdx.x;
  if (idx >= 6 * 2304) return;
  int l = idx / 2304, c = idx % 2304;
  float v = (c < 768) ? bq[l * 768 + c]
          : (c < 1536) ? bk[l * 768 + c - 768]
                       : bv[l * 768 + c - 1536];
  dst[idx] = v;
}

// ---------- atom encoder: h[b,0,:]=graph_token ; h[b,1+i,:] = x@W + b ----------
__global__ __launch_bounds__(256) void k_encode(const float* __restrict__ x,
    const float* __restrict__ W, const float* __restrict__ bias,
    const float* __restrict__ gt, float* __restrict__ h) {
  __shared__ float xs[64];
  int row = blockIdx.x, t = threadIdx.x;
  int b = row >> 8, tok = row & 255;
  float* hr = h + (size_t)row * 768;
  if (tok == 0) {                         // uniform per block
    for (int c = t; c < 768; c += 256) hr[c] = gt[c];
    return;
  }
  if (t < 64) xs[t] = x[((size_t)b * 255 + (tok - 1)) * 64 + t];
  __syncthreads();
  for (int c = t; c < 768; c += 256) {
    float a = bias[c];
#pragma unroll
    for (int k = 0; k < 64; ++k) a = fmaf(xs[k], W[k * 768 + c], a);
    hr[c] = a;
  }
}

// ---------- layernorm over 768; bf16 out (internal) or f32 out (final) ----------
__global__ __launch_bounds__(256) void k_layernorm(const float* __restrict__ x,
    const float* __restrict__ w, const float* __restrict__ bb,
    bf16* __restrict__ ob, float* __restrict__ of) {
  __shared__ float red[256];
  int row = blockIdx.x, t = threadIdx.x;
  const float* xr = x + (size_t)row * 768;
  float v0 = xr[t], v1 = xr[t + 256], v2 = xr[t + 512];
  red[t] = v0 + v1 + v2;
  __syncthreads();
  for (int o = 128; o > 0; o >>= 1) { if (t < o) red[t] += red[t + o]; __syncthreads(); }
  float mu = red[0] * (1.f / 768.f);
  __syncthreads();
  float d0 = v0 - mu, d1 = v1 - mu, d2 = v2 - mu;
  red[t] = d0 * d0 + d1 * d1 + d2 * d2;
  __syncthreads();
  for (int o = 128; o > 0; o >>= 1) { if (t < o) red[t] += red[t + o]; __syncthreads(); }
  float r = rsqrtf(red[0] * (1.f / 768.f) + 1e-5f);
  float y0 = d0 * r * w[t] + bb[t];
  float y1 = d1 * r * w[t + 256] + bb[t + 256];
  float y2 = d2 * r * w[t + 512] + bb[t + 512];
  if (ob) {
    bf16* orow = ob + (size_t)row * 768;
    orow[t] = (bf16)y0; orow[t + 256] = (bf16)y1; orow[t + 512] = (bf16)y2;
  } else {
    float* orow = of + (size_t)row * 768;
    orow[t] = y0; orow[t + 256] = y1; orow[t + 512] = y2;
  }
}

// ---------- WMMA GEMM, TDM-fed double-buffered A tile ----------
// Block: 256 thr (8 waves), tile 128(M) x 64(N), K-step 32.
// mode 0: outb = bf16(C) ; mode 1: resid += C (fp32) ; mode 2: outb = bf16(gelu(C))
__global__ __launch_bounds__(256) void k_gemm(const bf16* __restrict__ A,
    const bf16* __restrict__ Wsw, const float* __restrict__ bias,
    int M, int N, int K, int mode, float* __restrict__ resid,
    bf16* __restrict__ outb) {
  __shared__ bf16 As[2][128 * 32];
  int t = threadIdx.x, wave = t >> 5, lane = t & 31;
  int g = lane >> 4, ln = lane & 15;
  int m0 = blockIdx.y * 128, tn0 = blockIdx.x * 4;
  int ntiles = N >> 4;
  f32x8 acc[4];
#pragma unroll
  for (int tt = 0; tt < 4; ++tt)
#pragma unroll
    for (int i = 0; i < 8; ++i) acc[tt][i] = 0.f;
  int KC = K >> 5;
  u32 lds0 = (u32)(size_t)(void*)&As[0][0];
  const bf16* gtile = A + (size_t)m0 * K;

  if (wave == 0)                              // prime buffer 0 via TDM
    tdm_load_2d(lds0, gtile, 32, 128, (u32)K, 128, (u32)K);

  for (int ck = 0; ck < KC; ++ck) {
    if (wave == 0) {
      if (ck + 1 < KC) {                      // issue next tile, wait for current
        tdm_load_2d(lds0 + ((u32)((ck + 1) & 1)) * (128 * 32 * 2),
                    gtile + (size_t)(ck + 1) * 32, 32, 128, (u32)K, 128, (u32)K);
        __builtin_amdgcn_s_wait_tensorcnt((short)1);
      } else {
        __builtin_amdgcn_s_wait_tensorcnt((short)0);
      }
    }
    __syncthreads();   // single barrier per K-step (double buffer covers WAR)
    const bf16* al = As[ck & 1] + (wave * 16 + ln) * 32 + 16 * g;
    bf16x16 af = frag_from_pair(al, al + 8);
    const bf16* bp = Wsw + (((size_t)ck * ntiles + tn0) * 32 + lane) * 16;
    if (ck + 1 < KC)
      __builtin_prefetch(Wsw + (((size_t)(ck + 1) * ntiles + tn0) * 32 + lane) * 16, 0, 1);
    // materialize all 4 B fragments first: 8 global_load_b128 in flight,
    // then 4 back-to-back WMMAs retire them with partial loadcnt waits.
    bf16x16 bf0 = frag_contig(bp);
    bf16x16 bf1 = frag_contig(bp + 512);
    bf16x16 bf2 = frag_contig(bp + 1024);
    bf16x16 bf3 = frag_contig(bp + 1536);
    acc[0] = wmma_bf16(af, bf0, acc[0]);
    acc[1] = wmma_bf16(af, bf1, acc[1]);
    acc[2] = wmma_bf16(af, bf2, acc[2]);
    acc[3] = wmma_bf16(af, bf3, acc[3]);
  }

#pragma unroll
  for (int tt = 0; tt < 4; ++tt) {
    int nglob = blockIdx.x * 64 + tt * 16 + ln;
    float bv = bias ? bias[nglob] : 0.f;
#pragma unroll
    for (int vg = 0; vg < 8; ++vg) {
      int mrow = m0 + wave * 16 + vg + 8 * g;   // D layout: M = vg + 8*(lane/16)
      float v = acc[tt][vg] + bv;
      size_t o = (size_t)mrow * N + nglob;
      if (mode == 0) outb[o] = (bf16)v;
      else if (mode == 1) resid[o] += v;
      else { v = 0.5f * v * (1.f + erff(v * 0.70710678118f)); outb[o] = (bf16)v; }
    }
  }
}

// ---------- attention: one wave per (batch, head, 16-row tile) ----------
__global__ __launch_bounds__(32) void k_attn(const bf16* __restrict__ qkv,
    const int* __restrict__ spd, const float* __restrict__ emb,
    const float* __restrict__ virt, bf16* __restrict__ out) {
  __shared__ float S[16 * 256];
  __shared__ bf16  P[16 * 256];
  __shared__ bf16  Vt[32 * 64];
  int lane = threadIdx.x;
  int it = blockIdx.x & 15;
  int head = (blockIdx.x >> 4) % 12;
  int b = blockIdx.x / (16 * 12);
  int g = lane >> 4, ln = lane & 15;
  float vt = virt[head];
  const size_t rs = 2304;                  // qkv row stride
  const bf16* qbase = qkv + ((size_t)b * 256 + it * 16 + ln) * rs + head * 64;
  const bf16* kbase = qkv + ((size_t)b * 256 + ln) * rs + 768 + head * 64;

  // --- scores S = (Q K^T) * scale + bias ---
  for (int jt = 0; jt < 16; ++jt) {
    f32x8 acc;
#pragma unroll
    for (int i = 0; i < 8; ++i) acc[i] = 0.f;
    const bf16* kb = kbase + (size_t)jt * 16 * rs;
#pragma unroll
    for (int kc = 0; kc < 64; kc += 32) {
      bf16x16 qf = frag_from_pair(qbase + kc + 16 * g, qbase + kc + 16 * g + 8);
      bf16x16 kf = frag_contig(kb + kc + 16 * g);   // K^T B-frag: contiguous row
      acc = wmma_bf16(qf, kf, acc);
    }
#pragma unroll
    for (int vg = 0; vg < 8; ++vg) {
      int m = vg + 8 * g;
      int i = it * 16 + m;
      int j = jt * 16 + ln;
      float bias;
      if (i == 0 || j == 0) bias = vt;      // virtual-node distance rows/cols
      else {
        int sv = spd[((size_t)b * 255 + (i - 1)) * 255 + (j - 1)];
        bias = (sv == 0) ? 0.f : emb[sv * 12 + head];   // padding_idx = 0
      }
      S[m * 256 + j] = acc[vg] * 0.125f + bias;
    }
  }
  asm volatile("s_wait_dscnt 0x0" ::: "memory");

  // --- softmax over rows (wave32 shuffle reductions) ---
  for (int m = 0; m < 16; ++m) {
    float ev[8];
    float mx = -3.4e38f;
#pragma unroll
    for (int ci = 0; ci < 8; ++ci) mx = fmaxf(mx, S[m * 256 + lane + 32 * ci]);
#pragma unroll
    for (int off = 16; off > 0; off >>= 1) mx = fmaxf(mx, __shfl_xor(mx, off, 32));
    float sum = 0.f;
#pragma unroll
    for (int ci = 0; ci < 8; ++ci) {
      ev[ci] = __expf(S[m * 256 + lane + 32 * ci] - mx);
      sum += ev[ci];
    }
#pragma unroll
    for (int off = 16; off > 0; off >>= 1) sum += __shfl_xor(sum, off, 32);
    float inv = 1.f / sum;
#pragma unroll
    for (int ci = 0; ci < 8; ++ci)
      P[m * 256 + lane + 32 * ci] = (bf16)(ev[ci] * inv);
  }
  asm volatile("s_wait_dscnt 0x0" ::: "memory");

  // --- O = P @ V ---
  f32x8 oacc[4];
#pragma unroll
  for (int nt = 0; nt < 4; ++nt)
#pragma unroll
    for (int i = 0; i < 8; ++i) oacc[nt][i] = 0.f;
  for (int kt = 0; kt < 8; ++kt) {
    const bf16* vrow = qkv + ((size_t)b * 256 + kt * 32 + lane) * rs + 1536 + head * 64;
#pragma unroll
    for (int q = 0; q < 4; ++q)
      *(bf16x16*)(Vt + lane * 64 + q * 16) = *(const bf16x16*)(vrow + q * 16);
    asm volatile("s_wait_dscnt 0x0" ::: "memory");
    const bf16* pl = P + ln * 256 + kt * 32 + 16 * g;
    bf16x16 pf = frag_from_pair(pl, pl + 8);
#pragma unroll
    for (int nt = 0; nt < 4; ++nt) {
      bf16x16 vf;
#pragma unroll
      for (int e = 0; e < 16; ++e) vf[e] = Vt[(e + 16 * g) * 64 + nt * 16 + ln];
      oacc[nt] = wmma_bf16(pf, vf, oacc[nt]);
    }
  }
#pragma unroll
  for (int nt = 0; nt < 4; ++nt)
#pragma unroll
    for (int vg = 0; vg < 8; ++vg) {
      int i = it * 16 + vg + 8 * g;
      int col = head * 64 + nt * 16 + ln;
      out[((size_t)b * 256 + i) * 768 + col] = (bf16)oacc[nt][vg];
    }
}

// ---------------------------------------------------------------------------
extern "C" void kernel_launch(void* const* d_in, const int* in_sizes, int n_in,
                              void* d_out, int out_size, void* d_ws, size_t ws_size,
                              hipStream_t stream) {
  const float* x      = (const float*)d_in[0];
  const int*   spd    = (const int*)  d_in[1];
  const float* atomW  = (const float*)d_in[2];
  const float* atomB  = (const float*)d_in[3];
  const float* spdEmb = (const float*)d_in[4];
  const float* gtok   = (const float*)d_in[5];
  const float* virt   = (const float*)d_in[6];
  const float* ln1w   = (const float*)d_in[7];
  const float* ln1b   = (const float*)d_in[8];
  const float* Wq     = (const float*)d_in[9];
  const float* bq     = (const float*)d_in[10];
  const float* Wk     = (const float*)d_in[11];
  const float* bk     = (const float*)d_in[12];
  const float* Wv     = (const float*)d_in[13];
  const float* bv     = (const float*)d_in[14];
  const float* Wo     = (const float*)d_in[15];
  const float* bo     = (const float*)d_in[16];
  const float* ln2w   = (const float*)d_in[17];
  const float* ln2b   = (const float*)d_in[18];
  const float* W1     = (const float*)d_in[19];
  const float* b1     = (const float*)d_in[20];
  const float* W2     = (const float*)d_in[21];
  const float* b2     = (const float*)d_in[22];
  const float* flnw   = (const float*)d_in[23];
  const float* flnb   = (const float*)d_in[24];

  char* ws = (char*)d_ws;
  size_t off = 0;
  auto alloc = [&](size_t bytes) -> char* {
    char* p = ws + off;
    off = (off + bytes + 255) & ~(size_t)255;
    return p;
  };
  float* h    = (float*)alloc((size_t)8192 * 768 * 4);
  bf16*  y    = (bf16*) alloc((size_t)8192 * 768 * 2);
  bf16*  qkv  = (bf16*) alloc((size_t)8192 * 2304 * 2);
  bf16*  ao   = (bf16*) alloc((size_t)8192 * 768 * 2);
  bf16*  mid  = (bf16*) alloc((size_t)8192 * 3072 * 2);
  bf16*  wqkv = (bf16*) alloc((size_t)6 * 768 * 2304 * 2);
  bf16*  wo   = (bf16*) alloc((size_t)6 * 768 * 768 * 2);
  bf16*  w1   = (bf16*) alloc((size_t)6 * 768 * 3072 * 2);
  bf16*  w2   = (bf16*) alloc((size_t)6 * 3072 * 768 * 2);
  float* bqkv = (float*)alloc((size_t)6 * 2304 * 4);

  const int thr = 256;
  for (int l = 0; l < 6; ++l) {
    long kn = 768L * 768;
    k_swizzle<<<(kn + thr - 1) / thr, thr, 0, stream>>>(Wq + l * kn, wqkv + (size_t)l * 768 * 2304, 768, 768, 144, 0);
    k_swizzle<<<(kn + thr - 1) / thr, thr, 0, stream>>>(Wk + l * kn, wqkv + (size_t)l * 768 * 2304, 768, 768, 144, 48);
    k_swizzle<<<(kn + thr - 1) / thr, thr, 0, stream>>>(Wv + l * kn, wqkv + (size_t)l * 768 * 2304, 768, 768, 144, 96);
    k_swizzle<<<(kn + thr - 1) / thr, thr, 0, stream>>>(Wo + l * kn, wo + (size_t)l * 768 * 768, 768, 768, 48, 0);
    long kn1 = 768L * 3072;
    k_swizzle<<<(kn1 + thr - 1) / thr, thr, 0, stream>>>(W1 + l * kn1, w1 + (size_t)l * 768 * 3072, 768, 3072, 192, 0);
    k_swizzle<<<(kn1 + thr - 1) / thr, thr, 0, stream>>>(W2 + l * kn1, w2 + (size_t)l * 3072 * 768, 3072, 768, 48, 0);
  }
  k_pack_qkv_bias<<<(6 * 2304 + 255) / 256, 256, 0, stream>>>(bq, bk, bv, bqkv);
  k_encode<<<8192, 256, 0, stream>>>(x, atomW, atomB, gtok, h);

  for (int l = 0; l < 6; ++l) {
    k_layernorm<<<8192, 256, 0, stream>>>(h, ln1w + l * 768, ln1b + l * 768, y, nullptr);
    k_gemm<<<dim3(36, 64), 256, 0, stream>>>(y, wqkv + (size_t)l * 768 * 2304, bqkv + l * 2304,
                                             8192, 2304, 768, 0, nullptr, qkv);
    k_attn<<<32 * 12 * 16, 32, 0, stream>>>(qkv, spd, spdEmb, virt, ao);
    k_gemm<<<dim3(12, 64), 256, 0, stream>>>(ao, wo + (size_t)l * 768 * 768, bo + l * 768,
                                             8192, 768, 768, 1, h, nullptr);
    k_layernorm<<<8192, 256, 0, stream>>>(h, ln2w + l * 768, ln2b + l * 768, y, nullptr);
    k_gemm<<<dim3(48, 64), 256, 0, stream>>>(y, w1 + (size_t)l * 768 * 3072, b1 + l * 3072,
                                             8192, 3072, 768, 2, nullptr, mid);
    k_gemm<<<dim3(12, 64), 256, 0, stream>>>(mid, w2 + (size_t)l * 3072 * 768, b2 + l * 768,
                                             8192, 768, 3072, 1, h, nullptr);
  }
  k_layernorm<<<8192, 256, 0, stream>>>(h, flnw, flnb, nullptr, (float*)d_out);
  (void)in_sizes; (void)n_in; (void)out_size; (void)ws_size;
}
